// KnowledgeStorage_27264452395675
// MI455X (gfx1250) — matrix operand order repaired and checked by reading, not compiled, not run on hardware
//
#include <hip/hip_runtime.h>

// ---------------------------------------------------------------------------
// KnowledgeStorage for MI455X (gfx1250, wave32).
//
// Core matrix path: v_wmma_f32_16x16x32_bf16 (f32 inputs converted to bf16 in
// LDS, f32 accumulate). All large linears are flattened over the 40-way
// (batch x round) dimension into single GEMMs. Broadcast structure of the
// edge tensors is exploited so no (nc*nx, V+H) edge matrix is ever formed.
//
// GEMM kernel: 128 threads (4 waves), block tile 32(M) x 256(N), K step 32.
// Each wave: 2 A-fragments x 4 B-fragments -> 8 WMMAs per K step.
// Double-buffered LDS (1 barrier/iter), float4 staging loads, packed bf16
// LDS stores, global_prefetch for the K+2 weight tile.
//
// Assumed d_in flattening (recursive insertion order of setup_inputs()):
//   0 v_nodes (4,10,36,2048) f32     1 t_nodes (4,10,10,512) f32
//   2 ext_nodes (4,10,20,512) f32    3 ques_embed (4,10,512) f32
//   4 batch_size (1)                 5 num_rounds (1)
//   6.. params: modules in order [t2v, v2t, extk2v, extk2t], each with layers
//       ['4','5','b','6','c','l','7','e','8','a','9','lg','num_rounds'],
//       each layer as (w, b) -> 26 arrays per module.
// ---------------------------------------------------------------------------

typedef unsigned int  u32;
typedef unsigned short u16;

typedef __attribute__((ext_vector_type(16))) __bf16 v16bf;
typedef __attribute__((ext_vector_type(8)))  float  v8f;

union FragBF { v16bf v; u32 u[8]; };

__device__ __forceinline__ u16 f32_to_bf16(float f) {
  u32 u = __float_as_uint(f);
  u32 r = (u + 0x7FFFu + ((u >> 16) & 1u)) >> 16;   // round-to-nearest-even
  return (u16)r;
}
__device__ __forceinline__ u32 pack_bf2(float lo, float hi) {
  return (u32)f32_to_bf16(lo) | ((u32)f32_to_bf16(hi) << 16);
}

#define LDA_T 36   // u16 stride of LDS tile rows: 72B = 18 dwords ->
                   // conflict-free 16-lane gathers, 8B-aligned rows.

__device__ __forceinline__ float4 ld4_rowguard(const float* p, int rowOK,
                                               int c0, int Clim) {
  float4 v = {0.f, 0.f, 0.f, 0.f};
  if (rowOK) {
    if (c0 + 3 < Clim) {
      v = *(const float4*)p;
    } else {
      if (c0 < Clim)     v.x = p[0];
      if (c0 + 1 < Clim) v.y = p[1];
      if (c0 + 2 < Clim) v.z = p[2];
      if (c0 + 3 < Clim) v.w = p[3];
    }
  }
  return v;
}

// ---------------------------------------------------------------------------
// C[M,N] = act(A[M,K] @ W[K,N] + bias[N]);  act: 0 none, 1 sigmoid
// ---------------------------------------------------------------------------
__global__ __launch_bounds__(128) void ks_gemm_bf16(
    const float* __restrict__ A, int lda,
    const float* __restrict__ W, int ldw,
    const float* __restrict__ bias,
    float* __restrict__ C, int ldc,
    int M, int N, int K, int act)
{
  __shared__ u16 sA[2][32 * LDA_T];
  __shared__ u16 sW[2][256 * LDA_T];

  const int tid  = threadIdx.x;
  const int lane = tid & 31;
  const int wave = tid >> 5;
  const int m0   = blockIdx.y * 32;
  const int n0   = blockIdx.x * 256;

  const int halfsel = (lane >> 4) & 1;   // wave32 lane halves
  const int mlane   = lane & 15;

  v8f acc[2][4];
#pragma unroll
  for (int t = 0; t < 2; ++t)
#pragma unroll
    for (int s = 0; s < 4; ++s)
#pragma unroll
      for (int r = 0; r < 8; ++r) acc[t][s][r] = 0.0f;

  const int nk = (K + 31) / 32;

  // --- staging: tile kt*32 into buffer p (f32 -> bf16, transposed W) ---
  auto stage = [&](int p, int kt) {
    const int k0 = kt * 32;
    // A tile 32x32: float4 along K, packed b64 store (rows 72B -> 8B aligned)
    for (int idx = tid; idx < 32 * 8; idx += 128) {
      int mm = idx >> 3, kq = (idx & 7) * 4;
      int gm = m0 + mm, gk = k0 + kq;
      float4 v = ld4_rowguard(A + (size_t)gm * lda + gk, gm < M, gk, K);
      u32* dst = (u32*)&sA[p][mm * LDA_T + kq];
      dst[0] = pack_bf2(v.x, v.y);
      dst[1] = pack_bf2(v.z, v.w);
    }
    // W tile 32x256 -> transposed sW[n][k]; process K in pairs so each
    // (n,kpair) is one packed b32 store.
    for (int idx = tid; idx < 16 * 64; idx += 128) {
      int kp = (idx >> 6) * 2;
      int nq = (idx & 63) * 4;
      int gk = k0 + kp, gn = n0 + nq;
      float4 a = ld4_rowguard(W + (size_t)gk * ldw + gn,       gk < K,     gn, N);
      float4 b = ld4_rowguard(W + (size_t)(gk + 1) * ldw + gn, gk + 1 < K, gn, N);
      *(u32*)&sW[p][(nq + 0) * LDA_T + kp] = pack_bf2(a.x, b.x);
      *(u32*)&sW[p][(nq + 1) * LDA_T + kp] = pack_bf2(a.y, b.y);
      *(u32*)&sW[p][(nq + 2) * LDA_T + kp] = pack_bf2(a.z, b.z);
      *(u32*)&sW[p][(nq + 3) * LDA_T + kp] = pack_bf2(a.w, b.w);
    }
  };

  stage(0, 0);
  __syncthreads();

  int p = 0;
  for (int kt = 0; kt < nk; ++kt) {
    // prefetch W tile kt+2 (one cacheline per iteration per thread)
    if (kt + 2 < nk) {
      int k2 = (kt + 2) * 32;
      for (int i = tid; i < 256; i += 128) {        // 256 x 128B lines
        int row = i >> 3, col = (i & 7) * 32;
        if (k2 + row < K && n0 + col < N)
          __builtin_prefetch(W + (size_t)(k2 + row) * ldw + n0 + col, 0, 1);
      }
    }
    if (kt + 1 < nk) stage(p ^ 1, kt + 1);

    // --- fragments: load all, then 8 back-to-back WMMAs ---
    FragBF bfr[4];
#pragma unroll
    for (int s = 0; s < 4; ++s) {
      const u32* row = (const u32*)&sW[p][(wave * 64 + s * 16 + mlane) * LDA_T];
#pragma unroll
      for (int e = 0; e < 8; ++e) bfr[s].u[e] = row[halfsel * 8 + e];
    }
    FragBF afr[2];
#pragma unroll
    for (int t = 0; t < 2; ++t) {
      const u32* row = (const u32*)&sA[p][(t * 16 + mlane) * LDA_T];
#pragma unroll
      for (int v = 0; v < 4; ++v) {
        afr[t].u[v]     = row[halfsel * 8 + v];
        afr[t].u[4 + v] = row[halfsel * 8 + 4 + v];
      }
    }
#pragma unroll
    for (int t = 0; t < 2; ++t)
#pragma unroll
      for (int s = 0; s < 4; ++s)
        acc[t][s] = __builtin_amdgcn_wmma_f32_16x16x32_bf16(
            false, afr[t].v, false, bfr[s].v, (short)0, acc[t][s], false, false);

    __syncthreads();
    p ^= 1;
  }

  // --- store: subtile element r -> row m0 + t*16 + half*8 + r, col mlane ---
#pragma unroll
  for (int s = 0; s < 4; ++s) {
    int gn = n0 + wave * 64 + s * 16 + mlane;
    if (gn < N) {
      float bv = bias ? bias[gn] : 0.0f;
#pragma unroll
      for (int t = 0; t < 2; ++t) {
#pragma unroll
        for (int r = 0; r < 8; ++r) {
          int gm = m0 + t * 16 + halfsel * 8 + r;
          if (gm < M) {
            float v = acc[t][s][r] + bv;
            if (act == 1) v = 1.0f / (1.0f + __expf(-v));
            C[(size_t)gm * ldc + gn] = v;
          }
        }
      }
    }
  }
}

// ---------------------------------------------------------------------------
// Small helper kernels
// ---------------------------------------------------------------------------

// mask[row] = (sum_h |x[row,h]| > 0) ? 1 : 0
__global__ __launch_bounds__(256) void ks_rowmask(
    const float* __restrict__ x, float* __restrict__ mask, int rows, int D)
{
  int r = blockIdx.x * 256 + threadIdx.x;
  if (r >= rows) return;
  const float* p = x + (size_t)r * D;
  float s = 0.0f;
  for (int i = 0; i < D; ++i) s += fabsf(p[i]);
  mask[r] = (s > 0.0f) ? 1.0f : 0.0f;
}

// be[br,i,j,h] = mask ? cT[br,i,h] + xB[br,j,h] + b5[h] : b5[h]
__global__ __launch_bounds__(256) void ks_build_be(
    const float* __restrict__ cT, const float* __restrict__ xB,
    const float* __restrict__ mC, const float* __restrict__ mX,
    const float* __restrict__ b5, float* __restrict__ be,
    int total, int nc, int nx, int Hd)
{
  int idx = blockIdx.x * 256 + threadIdx.x;
  if (idx >= total) return;
  int h = idx % Hd;
  int t = idx / Hd;
  int j = t % nx; t /= nx;
  int i = t % nc;
  int br = t / nc;
  float m = mC[br * nc + i] * mX[br * nx + j];
  float v = b5[h];
  if (m > 0.0f)
    v += cT[(size_t)(br * nc + i) * Hd + h] + xB[(size_t)(br * nx + j) * Hd + h];
  be[idx] = v;
}

// out[row,h] = q[br,h] * in[row,h], br = row / rowsPerBr
__global__ __launch_bounds__(256) void ks_mul_qrow(
    const float* __restrict__ q, const float* __restrict__ in,
    float* __restrict__ out, int total, int Hd, int perBr)
{
  int idx = blockIdx.x * 256 + threadIdx.x;
  if (idx >= total) return;
  int br = idx / perBr;
  int h  = idx % Hd;
  out[idx] = q[(size_t)br * Hd + h] * in[idx];
}

// upd[br,i,j,h] = softmax_j(z[br,i,:,h]) * be[br,i,j,h]   (in-place ok)
__global__ __launch_bounds__(256) void ks_softmax_mul(
    const float* __restrict__ z, const float* __restrict__ be,
    float* __restrict__ out, int total, int nc, int nx, int Hd)
{
  int idx = blockIdx.x * 256 + threadIdx.x;
  if (idx >= total) return;          // total = BR*nc*Hd
  int h = idx % Hd;
  int t = idx / Hd;
  int i = t % nc;
  int br = t / nc;
  size_t base = ((size_t)(br * nc + i) * nx) * Hd + h;
  float mx = -3.0e38f;
  for (int j = 0; j < nx; ++j) mx = fmaxf(mx, z[base + (size_t)j * Hd]);
  float s = 0.0f;
  for (int j = 0; j < nx; ++j) s += __expf(z[base + (size_t)j * Hd] - mx);
  float inv = 1.0f / s;
  for (int j = 0; j < nx; ++j) {
    size_t o = base + (size_t)j * Hd;
    out[o] = __expf(z[o] - mx) * inv * be[o];
  }
}

// sb[br,i,j,h] += xb[br,j,h]
__global__ __launch_bounds__(256) void ks_add_bcast(
    float* __restrict__ sb, const float* __restrict__ xb,
    int total, int nc, int nx, int Hd)
{
  int idx = blockIdx.x * 256 + threadIdx.x;
  if (idx >= total) return;
  int h = idx % Hd;
  int t = idx / Hd;
  int j = t % nx; t /= nx;
  int br = t / nc;
  sb[idx] += xb[(size_t)(br * nx + j) * Hd + h];
}

// score[row] = bias + sum_h q[br,h]*s[row,h]*wv[h]; one wave32 per row.
__global__ __launch_bounds__(128) void ks_score_dot(
    const float* __restrict__ s, const float* __restrict__ q,
    const float* __restrict__ wv, const float* __restrict__ bp,
    float* __restrict__ out, int rows, int rowsPerBr, int Hd)
{
  int row = blockIdx.x * 4 + (threadIdx.x >> 5);
  int lane = threadIdx.x & 31;
  if (row >= rows) return;
  int br = row / rowsPerBr;
  const float* sp = s + (size_t)row * Hd;
  const float* qp = q + (size_t)br * Hd;
  float acc = 0.0f;
  for (int h = lane; h < Hd; h += 32) acc += qp[h] * sp[h] * wv[h];
#pragma unroll
  for (int off = 16; off > 0; off >>= 1) acc += __shfl_xor(acc, off, 32);
  if (lane == 0) out[row] = acc + bp[0];
}

// out[br,g,d] = sum_j softmax_j(sc[br,g,:]) * X[br,j,d]
__global__ __launch_bounds__(256) void ks_softmax_wsum(
    const float* __restrict__ sc, const float* __restrict__ X,
    float* __restrict__ out, int total, int G, int J, int D)
{
  int idx = blockIdx.x * 256 + threadIdx.x;
  if (idx >= total) return;          // total = BR*G*D
  int d = idx % D;
  int t = idx / D;
  int g = t % G;
  int br = t / G;
  const float* sp = sc + (size_t)(br * G + g) * J;
  float mx = -3.0e38f;
  for (int j = 0; j < J; ++j) mx = fmaxf(mx, sp[j]);
  float s = 0.0f;
  for (int j = 0; j < J; ++j) s += __expf(sp[j] - mx);
  float inv = 1.0f / s;
  const float* xp = X + (size_t)br * J * D + d;
  float a = 0.0f;
  for (int j = 0; j < J; ++j) a += __expf(sp[j] - mx) * inv * xp[(size_t)j * D];
  out[idx] = a;
}

// out[r, 0:Da] = a[r,:]; out[r, Da:Da+Db] = b[r,:]
__global__ __launch_bounds__(256) void ks_concat2(
    const float* __restrict__ a, int Da, const float* __restrict__ b, int Db,
    float* __restrict__ out, int total)
{
  int idx = blockIdx.x * 256 + threadIdx.x;
  if (idx >= total) return;
  int D = Da + Db;
  int c = idx % D;
  int r = idx / D;
  out[idx] = (c < Da) ? a[(size_t)r * Da + c] : b[(size_t)r * Db + (c - Da)];
}

// out[i] = a[i] * b[i]
__global__ __launch_bounds__(256) void ks_mul_ew(
    const float* __restrict__ a, const float* __restrict__ b,
    float* __restrict__ out, int total)
{
  int idx = blockIdx.x * 256 + threadIdx.x;
  if (idx >= total) return;
  out[idx] = a[idx] * b[idx];
}

// ---------------------------------------------------------------------------
// Host orchestration
// ---------------------------------------------------------------------------
static inline int cdiv(int a, int b) { return (a + b - 1) / b; }

extern "C" void kernel_launch(void* const* d_in, const int* in_sizes, int n_in,
                              void* d_out, int out_size, void* d_ws, size_t ws_size,
                              hipStream_t stream) {
  const float* v_nodes   = (const float*)d_in[0];
  const float* t_nodes   = (const float*)d_in[1];
  const float* ext_nodes = (const float*)d_in[2];
  const float* ques      = (const float*)d_in[3];

  const int Hd = 512, Vd = 2048;
  const int BR = in_sizes[3] / Hd;            // 40
  const int nv = in_sizes[0] / (BR * Vd);     // 36
  const int nt = in_sizes[1] / (BR * Hd);     // 10
  const int ne = in_sizes[2] / (BR * Hd);     // 20
  float* out = (float*)d_out;

  char* wsBase = (char*)d_ws;
  size_t wsOff = 0;
  auto alloc = [&](size_t nfloats) -> float* {
    float* p = (float*)(wsBase + wsOff);
    wsOff += ((nfloats * sizeof(float)) + 255) & ~(size_t)255;
    return p;
  };
  auto gemm = [&](const float* A, const float* W, const float* bias,
                  float* C, int M, int N, int K, int act) {
    dim3 g(cdiv(N, 256), cdiv(M, 32));
    ks_gemm_bf16<<<g, 128, 0, stream>>>(A, K, W, N, bias, C, N, M, N, K, act);
  };
  auto ewg = [&](int total) { return dim3(cdiv(total, 256)); };

  // Layer pair offsets within a module (w, b interleaved), layer order:
  // w4=0 b4=1 w5=2 b5=3 wb=4 bb=5 w6=6 b6=7 wc=8 bc=9 wl=10 bl=11 w7=12 b7=13
  // we=14 be=15 w8=16 b8=17 wa=18 ba=19 w9=20 b9=21 wlg=22 blg=23 wnr=24 bnr=25

  auto runModule = [&](const float* Cn, int nc, int Dc,     // center == storage nodes
                       const float* Xn, int nx, int Dx,     // cross nodes
                       int pbase, float* outPtr) {
    wsOff = 0;
    auto P = [&](int o) { return (const float*)d_in[pbase + o]; };
    const int E = nc * nx, Etot = BR * E;

    // ---- cross bridge ----
    float* q4 = alloc((size_t)BR * Hd);
    gemm(ques, P(0), P(1), q4, BR, Hd, Hd, 0);                       // q = lin(w4, ques)

    float* mC = alloc((size_t)BR * nc);
    float* mX = alloc((size_t)BR * nx);
    ks_rowmask<<<ewg(BR * nc), 256, 0, stream>>>(Cn, mC, BR * nc, Dc);
    ks_rowmask<<<ewg(BR * nx), 256, 0, stream>>>(Xn, mX, BR * nx, Dx);

    float* cT = alloc((size_t)BR * nc * Hd);                          // c @ W5_top
    gemm(Cn, P(2), nullptr, cT, BR * nc, Hd, Dc, 0);
    float* xB = alloc((size_t)BR * nx * Hd);                          // x @ W5_bot
    gemm(Xn, P(2) + (size_t)Dc * Hd, nullptr, xB, BR * nx, Hd, Dx, 0);

    float* be = alloc((size_t)Etot * Hd);                             // b_e
    ks_build_be<<<ewg(Etot * Hd), 256, 0, stream>>>(cT, xB, mC, mX, P(3), be,
                                                    Etot * Hd, nc, nx, Hd);

    float* qbe = alloc((size_t)Etot * Hd);                            // q * b_e
    ks_mul_qrow<<<ewg(Etot * Hd), 256, 0, stream>>>(q4, be, qbe, Etot * Hd, Hd, E * Hd);

    float* z = alloc((size_t)Etot * Hd);                              // lin(wb, q*b_e)
    gemm(qbe, P(4), P(5), z, Etot, Hd, Hd, 0);
    ks_softmax_mul<<<ewg(BR * nc * Hd), 256, 0, stream>>>(z, be, z,   // upd (in z)
                                                          BR * nc * Hd, nc, nx, Hd);

    float* xe6 = alloc((size_t)BR * nx * Hd);                         // x @ W6_top
    gemm(Xn, P(6), nullptr, xe6, BR * nx, Hd, Dx, 0);
    float* sb = qbe;                                                  // reuse buffer
    gemm(z, P(6) + (size_t)Dx * Hd, P(7), sb, Etot, Hd, Hd, 0);       // upd @ W6_bot + b6
    ks_add_bcast<<<ewg(Etot * Hd), 256, 0, stream>>>(sb, xe6, Etot * Hd, nc, nx, Hd);

    float* dsc = alloc((size_t)Etot);                                 // wc . (q*s_b) + bc
    ks_score_dot<<<cdiv(Etot, 4), 128, 0, stream>>>(sb, q4, P(8), P(9), dsc, Etot, E, Hd);

    float* u = alloc((size_t)BR * nc * Dx);                           // sum_j d * x
    ks_softmax_wsum<<<ewg(BR * nc * Dx), 256, 0, stream>>>(dsc, Xn, u,
                                                           BR * nc * Dx, nc, nx, Dx);

    // ---- storage (nodes == center) ----
    const int Dcat = Dc + Dx;
    float* cat = alloc((size_t)BR * nc * Dcat);
    ks_concat2<<<ewg(BR * nc * Dcat), 256, 0, stream>>>(Cn, Dc, u, Dx, cat, BR * nc * Dcat);
    float* gate = alloc((size_t)BR * nc * Dcat);
    gemm(cat, P(10), P(11), gate, BR * nc, Dcat, Dcat, 1);            // sigmoid gate
    ks_mul_ew<<<ewg(BR * nc * Dcat), 256, 0, stream>>>(gate, cat, gate, BR * nc * Dcat);
    float* local = alloc((size_t)BR * nc * Hd);
    gemm(gate, P(12), P(13), local, BR * nc, Hd, Dcat, 0);            // w7

    float* p8 = alloc((size_t)BR * nc * Hd);
    gemm(Cn, P(16), P(17), p8, BR * nc, Hd, Dc, 0);                   // w8(nodes)
    float* hs = alloc((size_t)BR * nc);
    ks_score_dot<<<cdiv(BR * nc, 4), 128, 0, stream>>>(p8, ques, P(14), P(15),
                                                       hs, BR * nc, nc, Hd);
    float* Ko = alloc((size_t)BR * Dc);
    ks_softmax_wsum<<<ewg(BR * Dc), 256, 0, stream>>>(hs, Cn, Ko, BR * Dc, 1, nc, Dc);

    gemm(local, P(20), P(21), p8, BR * nc, Hd, Hd, 0);                // w9(local) (reuse p8)
    ks_score_dot<<<cdiv(BR * nc, 4), 128, 0, stream>>>(p8, ques, P(18), P(19),
                                                       hs, BR * nc, nc, Hd);
    float* Kc = alloc((size_t)BR * Hd);
    ks_softmax_wsum<<<ewg(BR * Hd), 256, 0, stream>>>(hs, local, Kc, BR * Hd, 1, nc, Hd);

    const int Dk = Dc + Hd;
    float* Kcat = alloc((size_t)BR * Dk);
    ks_concat2<<<ewg(BR * Dk), 256, 0, stream>>>(Ko, Dc, Kc, Hd, Kcat, BR * Dk);
    float* g = alloc((size_t)BR * Dk);
    gemm(Kcat, P(22), P(23), g, BR, Dk, Dk, 1);                       // sigmoid lg
    ks_mul_ew<<<ewg(BR * Dk), 256, 0, stream>>>(g, Kcat, g, BR * Dk);
    gemm(g, P(24), P(25), outPtr, BR, Hd, Dk, 0);                     // num_rounds
  };

  // Param bases: modules stored in order t2v(6), v2t(32), extk2v(58), extk2t(84).
  // Output order: I_t, I_extk, H_i, H_extk.
  runModule(v_nodes, nv, Vd, t_nodes,   nt, Hd,  6, out + 0 * BR * Hd);  // I_t
  runModule(v_nodes, nv, Vd, ext_nodes, ne, Hd, 58, out + 1 * BR * Hd);  // I_extk
  runModule(t_nodes, nt, Hd, v_nodes,   nv, Vd, 32, out + 2 * BR * Hd);  // H_i
  runModule(t_nodes, nt, Hd, ext_nodes, ne, Hd, 84, out + 3 * BR * Hd);  // H_extk
}
